// Attention_22419729285910
// MI455X (gfx1250) — matrix operand order repaired and checked by reading, not compiled, not run on hardware
//
#include <hip/hip_runtime.h>
#include <hip/hip_bf16.h>

// ---- CDNA5 WMMA fragment types (wave32) ----
typedef __attribute__((ext_vector_type(16))) _Float16 v16h;
typedef __attribute__((ext_vector_type(8)))  _Float16 h8;
typedef __attribute__((ext_vector_type(8)))  float    v8f;
typedef __attribute__((vector_size(16)))     int      i4;   // 16B chunk for async copies

#define T_STEPS 16
#define BATCH   8
#define SEQ     196
#define CH      512
#define HEADS   8
#define HD      64
#define O3      1536                   // 3*CH
#define MROWS   (T_STEPS*BATCH*SEQ)    // 25088
#define SPATIAL (BATCH*SEQ*CH)         // 802816

// ---- gfx1250 async memory->LDS path (ASYNCcnt-tracked), with safe fallback
#if defined(__has_builtin)
#  if __has_builtin(__builtin_amdgcn_global_load_async_to_lds_b128) && \
      __has_builtin(__builtin_amdgcn_s_wait_asynccnt)
#    define USE_ASYNC_COPY 1
#  endif
#endif
#ifndef USE_ASYNC_COPY
#  define USE_ASYNC_COPY 0
#endif

__device__ __forceinline__ void copy16B(const _Float16* g, _Float16* l) {
#if USE_ASYNC_COPY
  __builtin_amdgcn_global_load_async_to_lds_b128(
      (__attribute__((address_space(1))) i4*)g,
      (__attribute__((address_space(3))) i4*)l, 0, 0);
#else
  *(h8*)l = *(const h8*)g;
#endif
}

__device__ __forceinline__ void wait_copies() {
#if USE_ASYNC_COPY
  __builtin_amdgcn_s_wait_asynccnt(0);
#endif
}

// -------------------------------------------------------------------------
// One-time f32 -> f16 weight conversion (weights then live in L2 as f16).
// -------------------------------------------------------------------------
__global__ void convert_w(const float* __restrict__ w, _Float16* __restrict__ wh, int n) {
  int i = blockIdx.x * blockDim.x + threadIdx.x;
  if (i < n) wh[i] = (_Float16)w[i];
}

// -------------------------------------------------------------------------
// LIF neuron: sequential scan over T per spatial location.
// -------------------------------------------------------------------------
__global__ void lif_scan(const float* __restrict__ x, _Float16* __restrict__ s,
                         const float* __restrict__ w, int spatial) {
  int i = blockIdx.x * blockDim.x + threadIdx.x;
  if (i >= spatial) return;
  float sg = 1.f / (1.f + __expf(-w[0]));
  float v = 0.f;
  for (int t = 0; t < T_STEPS; ++t) {
    size_t idx = (size_t)t * spatial + i;
    v += (x[idx] - v) * sg;
    float sp = (v >= 0.5f) ? 1.f : 0.f;
    v *= (1.f - sp);
    s[idx] = (_Float16)sp;
  }
}

// -------------------------------------------------------------------------
// Spike GEMM: out[m,o] = sum_c A[m,c]*W[o,c] + bias[o]
// A: [M,K] f16 spikes, Wh: [O,K] f16, out f32.
// Block tile 32(M) x 256(O), 8 waves; wave owns a 16x64 strip = 4 WMMA accs.
// Double-buffered LDS; next tile copied via GLOBAL_LOAD_ASYNC_TO_LDS_B128
// while current tile feeds v_wmma. One s_wait_asynccnt(0)+barrier per step.
// -------------------------------------------------------------------------
__global__ void gemm_spike(const _Float16* __restrict__ A,
                           const _Float16* __restrict__ Wh,
                           const float* __restrict__ bias,
                           float* __restrict__ Cout,
                           int M, int K, int O) {
  __shared__ __attribute__((aligned(16))) _Float16 As[2][32][40];    // 80B stride
  __shared__ __attribute__((aligned(16))) _Float16 Bs[2][256][40];
  int tid = threadIdx.x, lane = tid & 31, wave = tid >> 5;
  int hs = lane >> 4, l16 = lane & 15;
  int m0 = blockIdx.x * 32, n0 = blockIdx.y * 256;
  int wm = (wave >> 2) * 16;             // 0 / 16
  int wn = (wave & 3) * 64;              // 0 / 64 / 128 / 192

  auto issue = [&](int p, int k0) {
    if (tid < 128) {                     // A tile: 32x32 halves (128 x 16B)
      int r = tid >> 2, cc = (tid & 3) * 8;
      copy16B(&A[(size_t)(m0 + r) * K + k0 + cc], &As[p][r][cc]);
    }
    #pragma unroll
    for (int i = 0; i < 4; ++i) {        // B tile: 256 cols x 32 k (1024 x 16B)
      int chunk = tid + i * 256;
      int col = chunk >> 2, kc = (chunk & 3) * 8;
      copy16B(&Wh[(size_t)(n0 + col) * K + k0 + kc], &Bs[p][col][kc]);
    }
  };

  v8f acc[4] = {};
  const int NK = K >> 5;
  issue(0, 0);
  for (int it = 0; it < NK; ++it) {
    int p = it & 1;
    wait_copies();                       // own async copies into buffer p done
    __syncthreads();                     // all waves' copies done + prev compute done
    if (it + 1 < NK) issue(p ^ 1, (it + 1) * 32);

    v16h a; h8* ah = (h8*)&a;
    ah[0] = *(const h8*)&As[p][wm + l16][hs * 8];
    ah[1] = *(const h8*)&As[p][wm + l16][16 + hs * 8];
    #pragma unroll
    for (int j = 0; j < 4; ++j) {
      v16h bv; h8* bh = (h8*)&bv;
      bh[0] = *(const h8*)&Bs[p][wn + j * 16 + l16][hs * 8];
      bh[1] = *(const h8*)&Bs[p][wn + j * 16 + l16][16 + hs * 8];
      acc[j] = __builtin_amdgcn_wmma_f32_16x16x32_f16(false, a, false, bv,
                                                      (short)0, acc[j], false, false);
    }
  }

  #pragma unroll
  for (int j = 0; j < 4; ++j) {
    float* aj = (float*)&acc[j];
    #pragma unroll
    for (int r = 0; r < 8; ++r) {
      int mrow = m0 + wm + hs * 8 + r;
      int oc = n0 + wn + j * 16 + l16;
      Cout[(size_t)mrow * O + oc] = aj[r] + bias[oc];
    }
  }
}

// -------------------------------------------------------------------------
// BatchNorm stats, coalesced: one block per (t, 64-channel group).
// -------------------------------------------------------------------------
__global__ void bn_stats(const float* __restrict__ x, float* __restrict__ mean,
                         float* __restrict__ rstd, int O) {
  int groups = O >> 6;
  int t = blockIdx.x / groups, o0 = (blockIdx.x % groups) * 64;
  int oc = threadIdx.x & 63, sl = threadIdx.x >> 6;       // 4 slices
  __shared__ float s1[256], s2[256];
  float a = 0.f, b = 0.f;
  for (int i = sl; i < BATCH * SEQ; i += 4) {
    float v = x[((size_t)t * BATCH * SEQ + i) * O + o0 + oc];
    a += v; b += v * v;
  }
  s1[threadIdx.x] = a; s2[threadIdx.x] = b;
  __syncthreads();
  if (threadIdx.x < 64) {
    float sa = s1[threadIdx.x] + s1[threadIdx.x + 64] + s1[threadIdx.x + 128] + s1[threadIdx.x + 192];
    float sb = s2[threadIdx.x] + s2[threadIdx.x + 64] + s2[threadIdx.x + 128] + s2[threadIdx.x + 192];
    float m = sa / (float)(BATCH * SEQ);
    float var = sb / (float)(BATCH * SEQ) - m * m;
    mean[t * O + o0 + oc] = m;
    rstd[t * O + o0 + oc] = rsqrtf(var + 1e-5f);
  }
}

// -------------------------------------------------------------------------
// BN1 apply + q/k/v LIF (fused): normalize qkv, then per-stream LIF scan.
// -------------------------------------------------------------------------
__global__ void bn_lif_qkv(const float* __restrict__ qkv,
                           const float* __restrict__ mean, const float* __restrict__ rstd,
                           const float* __restrict__ g, const float* __restrict__ be,
                           const float* __restrict__ wq, const float* __restrict__ wk,
                           const float* __restrict__ wv, _Float16* __restrict__ s) {
  int i = blockIdx.x * blockDim.x + threadIdx.x;   // over B*N*O3
  if (i >= BATCH * SEQ * O3) return;
  int o = i % O3; int bn = i / O3;
  int stream = o >> 9;                             // 0=q, 1=k, 2=v
  float w = (stream == 0) ? wq[0] : (stream == 1 ? wk[0] : wv[0]);
  float sg = 1.f / (1.f + __expf(-w));
  float v = 0.f;
  for (int t = 0; t < T_STEPS; ++t) {
    size_t idx = ((size_t)t * BATCH * SEQ + bn) * O3 + o;
    int ch = t * O3 + o;
    float val = (qkv[idx] - mean[ch]) * rstd[ch] * g[ch] + be[ch];
    v += (val - v) * sg;
    float sp = (v >= 0.5f) ? 1.f : 0.f;
    v *= (1.f - sp);
    s[idx] = (_Float16)sp;
  }
}

// -------------------------------------------------------------------------
// Attention per (t,b,h): kv = k^T v (64x64, K=196 padded), out = q*kv*scale.
// k/v staged TRANSPOSED [d][n] so both fragments are contiguous b128 loads;
// kv parked transposed [e][d] (aligned h8 stores) for contiguous phase-2 B.
// -------------------------------------------------------------------------
__global__ void attention(const _Float16* __restrict__ qkv_s,
                          float* __restrict__ out) {
  int blk = blockIdx.x;                  // t*64 + b*8 + h
  int h = blk & 7, b = (blk >> 3) & 7, t = blk >> 6;
  size_t base = (size_t)(t * BATCH + b) * SEQ * O3;
  int qoff = h * HD, koff = CH + h * HD, voff = 2 * CH + h * HD;

  __shared__ __attribute__((aligned(16))) _Float16 ksht[64][40];   // [d][n]
  __shared__ __attribute__((aligned(16))) _Float16 vsht[64][40];   // [d][n]
  __shared__ __attribute__((aligned(16))) _Float16 kvsh[64][72];   // [e][d]

  int tid = threadIdx.x;                 // 128 = 4 waves
  int lane = tid & 31, wave = tid >> 5;
  int hs = lane >> 4, l16 = lane & 15;

  // -------- phase 1: kv = k^T @ v, accumulate over N in chunks of 32 --------
  v8f acc[4] = {};
  for (int nb = 0; nb < 7; ++nb) {       // 7*32 = 224 >= 196 (zero padded)
    int n0 = nb * 32;
    #pragma unroll
    for (int i = 0; i < 2; ++i) {        // vector global reads, transposed LDS stores
      int idx = tid + i * 128;           // 0..255
      int nl = idx >> 3;                 // 0..31
      int dc = (idx & 7) * 8;            // 0..56
      int n = n0 + nl;
      h8 kv8 = {}, vv8 = {};
      if (n < SEQ) {
        size_t rb = base + (size_t)n * O3;
        kv8 = *(const h8*)&qkv_s[rb + koff + dc];
        vv8 = *(const h8*)&qkv_s[rb + voff + dc];
      }
      #pragma unroll
      for (int e = 0; e < 8; ++e) { ksht[dc + e][nl] = kv8[e]; vsht[dc + e][nl] = vv8[e]; }
    }
    __syncthreads();

    // A = k^T: row d = wave*16+l16, K=n contiguous in ksht row
    v16h a; h8* ah = (h8*)&a;
    ah[0] = *(const h8*)&ksht[wave * 16 + l16][hs * 8];
    ah[1] = *(const h8*)&ksht[wave * 16 + l16][16 + hs * 8];
    #pragma unroll
    for (int j = 0; j < 4; ++j) {        // B = v: col e = j*16+l16, K=n contiguous
      v16h bv; h8* bh = (h8*)&bv;
      bh[0] = *(const h8*)&vsht[j * 16 + l16][hs * 8];
      bh[1] = *(const h8*)&vsht[j * 16 + l16][16 + hs * 8];
      acc[j] = __builtin_amdgcn_wmma_f32_16x16x32_f16(false, a, false, bv,
                                                      (short)0, acc[j], false, false);
    }
    __syncthreads();
  }
  // park kv transposed [e][d] (values are integer counts <= 196: exact in f16)
  #pragma unroll
  for (int j = 0; j < 4; ++j) {
    float* aj = (float*)&acc[j];
    h8 tmp;
    #pragma unroll
    for (int r = 0; r < 8; ++r) tmp[r] = (_Float16)aj[r];
    *(h8*)&kvsh[j * 16 + l16][wave * 16 + hs * 8] = tmp;   // 144B row stride: 16B aligned
  }
  __syncthreads();

  // -------- phase 2: out[n,e] = scale * q[n,:] @ kv, n tiled by 16 --------
  const float scale = 0.125f;            // hd^-0.5
  for (int nt = wave; nt < 13; nt += 4) {
    v8f oacc[4] = {};
    #pragma unroll
    for (int kb = 0; kb < 2; ++kb) {     // K = 64 in two steps of 32
      v16h a2; h8* ah2 = (h8*)&a2;
      int n = nt * 16 + l16;
      if (n < SEQ) {
        size_t rb = base + (size_t)n * O3 + qoff + kb * 32;
        ah2[0] = *(const h8*)&qkv_s[rb + hs * 8];
        ah2[1] = *(const h8*)&qkv_s[rb + 16 + hs * 8];
      } else {
        h8 z = {}; ah2[0] = z; ah2[1] = z;
      }
      #pragma unroll
      for (int j = 0; j < 4; ++j) {      // B = kv: col e = j*16+l16, K=d contiguous
        v16h bv; h8* bh = (h8*)&bv;
        bh[0] = *(const h8*)&kvsh[j * 16 + l16][kb * 32 + hs * 8];
        bh[1] = *(const h8*)&kvsh[j * 16 + l16][kb * 32 + 16 + hs * 8];
        oacc[j] = __builtin_amdgcn_wmma_f32_16x16x32_f16(false, a2, false, bv,
                                                         (short)0, oacc[j], false, false);
      }
    }
    #pragma unroll
    for (int j = 0; j < 4; ++j) {
      float* op = (float*)&oacc[j];
      #pragma unroll
      for (int r = 0; r < 8; ++r) {
        int n = nt * 16 + hs * 8 + r;
        if (n < SEQ)
          out[((size_t)(t * BATCH + b) * SEQ + n) * CH + h * HD + j * 16 + l16] = op[r] * scale;
      }
    }
  }
}

// -------------------------------------------------------------------------
// BN2 apply -> final output [T,B,N,C] f32.
// -------------------------------------------------------------------------
__global__ void bn_apply(const float* __restrict__ x,
                         const float* __restrict__ mean, const float* __restrict__ rstd,
                         const float* __restrict__ g, const float* __restrict__ be,
                         float* __restrict__ out) {
  size_t i = (size_t)blockIdx.x * blockDim.x + threadIdx.x;
  const size_t total = (size_t)T_STEPS * BATCH * SEQ * CH;
  if (i >= total) return;
  int o = (int)(i % CH);
  int t = (int)(i / ((size_t)BATCH * SEQ * CH));
  int ch = t * CH + o;
  out[i] = (x[i] - mean[ch]) * rstd[ch] * g[ch] + be[ch];
}

// -------------------------------------------------------------------------
extern "C" void kernel_launch(void* const* d_in, const int* in_sizes, int n_in,
                              void* d_out, int out_size, void* d_ws, size_t ws_size,
                              hipStream_t stream) {
  const float* x      = (const float*)d_in[0];
  const float* qkv_w  = (const float*)d_in[1];
  const float* qkv_b  = (const float*)d_in[2];
  const float* bn1_g  = (const float*)d_in[3];
  const float* bn1_b  = (const float*)d_in[4];
  const float* proj_w = (const float*)d_in[5];
  const float* proj_b = (const float*)d_in[6];
  const float* bn2_g  = (const float*)d_in[7];
  const float* bn2_b  = (const float*)d_in[8];
  const float* w_in   = (const float*)d_in[9];
  const float* w_q    = (const float*)d_in[10];
  const float* w_k    = (const float*)d_in[11];
  const float* w_v    = (const float*)d_in[12];
  const float* w_proj = (const float*)d_in[13];

  const size_t TOT = (size_t)T_STEPS * SPATIAL;
  char* p = (char*)d_ws;
  _Float16* s_in    = (_Float16*)p;  p += TOT * 2;                      // input spikes
  float*    qkv     = (float*)p;     p += (size_t)MROWS * O3 * 4;       // qkv pre-BN
  float*    bn1_m   = (float*)p;     p += (size_t)T_STEPS * O3 * 4;
  float*    bn1_r   = (float*)p;     p += (size_t)T_STEPS * O3 * 4;
  _Float16* qkv_s   = (_Float16*)p;  p += (size_t)MROWS * O3 * 2;       // q/k/v spikes
  float*    attn    = (float*)p;     p += TOT * 4;                      // attention out
  _Float16* s_attn  = (_Float16*)p;  p += TOT * 2;                      // proj-input spikes
  float*    proj    = (float*)p;     p += TOT * 4;                      // proj pre-BN
  float*    bn2_m   = (float*)p;     p += (size_t)T_STEPS * CH * 4;
  float*    bn2_r   = (float*)p;     p += (size_t)T_STEPS * CH * 4;
  _Float16* qkv_wh  = (_Float16*)p;  p += (size_t)O3 * CH * 2;          // f16 weights
  _Float16* proj_wh = (_Float16*)p;  p += (size_t)CH * CH * 2;

  // 0. one-time weight conversion to f16
  convert_w<<<(O3 * CH + 255) / 256, 256, 0, stream>>>(qkv_w, qkv_wh, O3 * CH);
  convert_w<<<(CH * CH + 255) / 256, 256, 0, stream>>>(proj_w, proj_wh, CH * CH);
  // 1. input LIF -> spikes
  lif_scan<<<(SPATIAL + 255) / 256, 256, 0, stream>>>(x, s_in, w_in, SPATIAL);
  // 2. qkv = spikes @ qkv_w^T + b   (WMMA f16, async double-buffered)
  gemm_spike<<<dim3(MROWS / 32, O3 / 256), 256, 0, stream>>>(s_in, qkv_wh, qkv_b, qkv,
                                                             MROWS, CH, O3);
  // 3. BN1 stats per (t, 3C-channel), coalesced
  bn_stats<<<T_STEPS * (O3 / 64), 256, 0, stream>>>(qkv, bn1_m, bn1_r, O3);
  // 4. BN1 apply + q/k/v LIF
  bn_lif_qkv<<<(BATCH * SEQ * O3 + 255) / 256, 256, 0, stream>>>(
      qkv, bn1_m, bn1_r, bn1_g, bn1_b, w_q, w_k, w_v, qkv_s);
  // 5. attention per (t,b,h)  (WMMA f16)
  attention<<<T_STEPS * BATCH * HEADS, 128, 0, stream>>>(qkv_s, attn);
  // 6. proj LIF
  lif_scan<<<(SPATIAL + 255) / 256, 256, 0, stream>>>(attn, s_attn, w_proj, SPATIAL);
  // 7. proj GEMM  (WMMA f16, async double-buffered)
  gemm_spike<<<dim3(MROWS / 32, CH / 256), 256, 0, stream>>>(s_attn, proj_wh, proj_b, proj,
                                                             MROWS, CH, CH);
  // 8. BN2 stats
  bn_stats<<<T_STEPS * (CH / 64), 256, 0, stream>>>(proj, bn2_m, bn2_r, CH);
  // 9. BN2 apply -> d_out
  bn_apply<<<(int)((TOT + 255) / 256), 256, 0, stream>>>(proj, bn2_m, bn2_r,
                                                         bn2_g, bn2_b, (float*)d_out);
}
